// LSTM_21431886807587
// MI455X (gfx1250) — matrix operand order repaired and checked by reading, not compiled, not run on hardware
//
#include <hip/hip_runtime.h>

typedef __attribute__((ext_vector_type(16))) __bf16 v16bf;
typedef __attribute__((ext_vector_type(8)))  __bf16 v8bf;
typedef __attribute__((ext_vector_type(8)))  float  v8f;
typedef __attribute__((ext_vector_type(4)))  float  v4f;

#define S_LEN 2048
#define IN    16
#define HID   64
#define ROWS  16        // batch rows per workgroup (one M tile)
#define NWAVE 16        // one wave per 16-wide gate-column tile (4H=256)
#define NTHREADS (NWAVE * 32)
#define ASTR  104       // bf16 row stride of A panel (16B aligned, conflict-free)
#define GSTR  20        // f32 stride of TRANSPOSED gate panel: G[col*20 + row]
#define GPAN  (256 * GSTR)   // 5120 floats

// ---- fast transcendentals: keep the serial path on TRANS ops only ----
#if __has_builtin(__builtin_amdgcn_tanhf)
__device__ __forceinline__ float fast_tanh(float x) {
    return __builtin_amdgcn_tanhf(x);           // v_tanh_f32 (gfx1250)
}
__device__ __forceinline__ float fast_sigmoid(float x) {
    return __builtin_amdgcn_tanhf(x * 0.5f) * 0.5f + 0.5f;
}
#else
#define LOG2E 1.4426950408889634f
__device__ __forceinline__ float fast_sigmoid(float x) {
    return __builtin_amdgcn_rcpf(1.0f + __builtin_amdgcn_exp2f(-x * LOG2E));
}
__device__ __forceinline__ float fast_tanh(float x) {
    return 2.0f * __builtin_amdgcn_rcpf(1.0f + __builtin_amdgcn_exp2f(-2.0f * LOG2E * x)) - 1.0f;
}
#endif

__global__ __launch_bounds__(NTHREADS)
void lstm_scan_wmma(const float* __restrict__ x,
                    const float* __restrict__ W_ih,
                    const float* __restrict__ W_hh,
                    const float* __restrict__ b_ih,
                    const float* __restrict__ b_hh,
                    const float* __restrict__ W1,
                    const float* __restrict__ b1,
                    const float* __restrict__ W2,
                    const float* __restrict__ b2,
                    float* __restrict__ out)
{
    // A panel: 16 batch rows x [x(16) | h(64) | zero-pad] bf16
    __shared__ __align__(16) __bf16 Abuf[ROWS * ASTR];
    // Transposed gate panel G[col][row] (also reused by the head)
    __shared__ __align__(16) float  Gpan[GPAN];

    const int t    = threadIdx.x;
    const int lane = t & 31;
    const int w    = t >> 5;                 // wave id == gate column tile
    const int b0   = blockIdx.x * ROWS;

    // ---- zero A staging (also sets h0 = 0 and K-pad = 0) ----
    for (int i = t; i < ROWS * ASTR; i += NTHREADS) Abuf[i] = (__bf16)0.0f;

    // ---- resident per-wave B fragments: Wcat^T tile [96 x 16] in bf16 ----
    // B fragment k order used here (contiguous, same as A): lane(0..15) = column N, lanes 16..31 K+=16,
    // VGPR v holds K = 2v, 2v+1 in halves.
    const int ncol = w * 16 + (lane & 15);
    const int khi  = (lane >= 16) ? 16 : 0;
    v16bf bfrag[3];
    #pragma unroll
    for (int kk = 0; kk < 3; ++kk) {
        #pragma unroll
        for (int v = 0; v < 8; ++v) {
            #pragma unroll
            for (int h = 0; h < 2; ++h) {
                int k = kk * 32 + khi + 2 * v + h;
                float wv;
                if (k < IN)            wv = W_ih[ncol * IN + k];
                else if (k < IN + HID) wv = W_hh[ncol * HID + (k - IN)];
                else                   wv = 0.0f;
                bfrag[kk][2 * v + h] = (__bf16)wv;
            }
        }
    }
    const float bias = b_ih[ncol] + b_hh[ncol];   // folded into accumulator
    v8f cbias;
    #pragma unroll
    for (int r = 0; r < 8; ++r) cbias[r] = bias;

    // ---- preload x_0 into A cols [0,16) ----
    if (t < 128) {
        int r = t >> 3, i2 = (t & 7) * 2;
        const float* xp = &x[((size_t)(b0 + r) * S_LEN) * IN + i2];
        union { __bf16 h[2]; unsigned u; } p;
        p.h[0] = (__bf16)xp[0]; p.h[1] = (__bf16)xp[1];
        *(unsigned*)&Abuf[r * ASTR + i2] = p.u;
    }
    __syncthreads();

    // elementwise ownership: cell (m, j), (m, j+1); bank-conflict-free reads
    const int em = t & 15;                    // batch row
    const int ej = (t >> 4) * 2;              // hidden column pair
    float c0 = 0.0f, c1 = 0.0f;
    float h0f = 0.0f, h1f = 0.0f;

    // A-fragment LDS addressing (16-bit A 16x32 layout)
    const int arow  = (lane & 15) * ASTR;
    const int aoff1 = (lane >= 16) ? 16 : 0;

    // gate-tile store addressing (transposed panel)
    const int gbase = (w * 16 + (lane & 15)) * GSTR + ((lane >= 16) ? 8 : 0);

    const int xr  = t >> 3;                   // x staging: row, col pair
    const int xi2 = (t & 7) * 2;

    #pragma unroll 1
    for (int s = 0; s < S_LEN; ++s) {
        // ---- phase A: gates = [x_t | h] * Wcat^T + bias via WMMA ----
        v8bf ck[6];
        #pragma unroll
        for (int kk = 0; kk < 3; ++kk) {
            const v8bf* ap = (const v8bf*)&Abuf[arow + kk * 32 + aoff1];
            ck[2 * kk]     = ap[0];          // K +0..7
            ck[2 * kk + 1] = ap[1];          // K +8..15
        }
        v8f acc = cbias;
        #pragma unroll
        for (int kk = 0; kk < 3; ++kk) {
            v16bf afrag = __builtin_shufflevector(ck[2 * kk], ck[2 * kk + 1],
                0, 1, 2, 3, 4, 5, 6, 7, 8, 9, 10, 11, 12, 13, 14, 15);
            acc = __builtin_amdgcn_wmma_f32_16x16x32_bf16(
                false, afrag, false, bfrag[kk], (short)0, acc, false, false);
        }
        // transposed gate-tile store: lane's 8 values are consecutive ->
        // two ds_store_b128
        {
            v4f lo, hi;
            #pragma unroll
            for (int r = 0; r < 4; ++r) { lo[r] = acc[r]; hi[r] = acc[4 + r]; }
            *(v4f*)&Gpan[gbase]     = lo;
            *(v4f*)&Gpan[gbase + 4] = hi;
        }
        __syncthreads();

        // ---- phase B ----
        // issue next-x global loads first so VMEM overlaps the elementwise math
        float xn0 = 0.0f, xn1 = 0.0f;
        const bool do_x = (s + 1 < S_LEN) && (t < 128);
        if (do_x) {
            const float* xp = &x[((size_t)(b0 + xr) * S_LEN + (s + 1)) * IN + xi2];
            xn0 = xp[0];
            xn1 = xp[1];
            if (s + 2 < S_LEN)
                __builtin_prefetch(&x[((size_t)(b0 + xr) * S_LEN + (s + 2)) * IN + xi2], 0, 3);
        }

        // LSTM cell elementwise (c in registers), TRANS-only nonlinearities
        {
            float iv0 = fast_sigmoid(Gpan[(ej)            * GSTR + em]);
            float iv1 = fast_sigmoid(Gpan[(ej + 1)        * GSTR + em]);
            float fv0 = fast_sigmoid(Gpan[(HID + ej)      * GSTR + em]);
            float fv1 = fast_sigmoid(Gpan[(HID + ej + 1)  * GSTR + em]);
            float gv0 = fast_tanh   (Gpan[(2*HID + ej)    * GSTR + em]);
            float gv1 = fast_tanh   (Gpan[(2*HID + ej + 1)* GSTR + em]);
            float ov0 = fast_sigmoid(Gpan[(3*HID + ej)    * GSTR + em]);
            float ov1 = fast_sigmoid(Gpan[(3*HID + ej + 1)* GSTR + em]);
            c0 = fv0 * c0 + iv0 * gv0;
            c1 = fv1 * c1 + iv1 * gv1;
            h0f = ov0 * fast_tanh(c0);
            h1f = ov1 * fast_tanh(c1);
            union { __bf16 h[2]; unsigned u; } p;
            p.h[0] = (__bf16)h0f; p.h[1] = (__bf16)h1f;
            *(unsigned*)&Abuf[em * ASTR + IN + ej] = p.u;   // new h into A panel
        }
        if (do_x) {
            union { __bf16 h[2]; unsigned u; } p;
            p.h[0] = (__bf16)xn0; p.h[1] = (__bf16)xn1;
            *(unsigned*)&Abuf[xr * ASTR + xi2] = p.u;       // x_{s+1} into A panel
        }
        __syncthreads();
    }

    // ---- head: y = (h @ W1^T + b1) @ W2^T + b2 (f32, negligible) ----
    // stash f32 h at Gpan[m*68 + j]; y1 at Gpan[1536 + r*68 + k]
    Gpan[em * 68 + ej]     = h0f;
    Gpan[em * 68 + ej + 1] = h1f;
    __syncthreads();

    {   // y1: 16 rows x 64, two outputs per thread
        int r = t >> 5, k = (t & 31) * 2;
        float a0 = b1[k], a1 = b1[k + 1];
        const float* hr = &Gpan[r * 68];
        #pragma unroll 8
        for (int j = 0; j < HID; ++j) {
            float hv = hr[j];
            a0 += hv * W1[k * HID + j];
            a1 += hv * W1[(k + 1) * HID + j];
        }
        Gpan[1536 + r * 68 + k]     = a0;
        Gpan[1536 + r * 68 + k + 1] = a1;
    }
    __syncthreads();

    if (t < ROWS * 9) {
        int r = t / 9, o = t % 9;
        float a = b2[o];
        const float* yr = &Gpan[1536 + r * 68];
        #pragma unroll 8
        for (int k = 0; k < HID; ++k)
            a += yr[k] * W2[o * HID + k];
        out[(size_t)(b0 + r) * 9 + o] = a;
    }
}

extern "C" void kernel_launch(void* const* d_in, const int* in_sizes, int n_in,
                              void* d_out, int out_size, void* d_ws, size_t ws_size,
                              hipStream_t stream) {
    (void)n_in; (void)d_ws; (void)ws_size; (void)out_size;
    const float* x    = (const float*)d_in[0];
    const float* W_ih = (const float*)d_in[1];
    const float* W_hh = (const float*)d_in[2];
    const float* b_ih = (const float*)d_in[3];
    const float* b_hh = (const float*)d_in[4];
    const float* W1   = (const float*)d_in[5];
    const float* b1   = (const float*)d_in[6];
    const float* W2   = (const float*)d_in[7];
    const float* b2   = (const float*)d_in[8];
    float* out = (float*)d_out;

    const int B = in_sizes[0] / (S_LEN * IN);   // 512
    dim3 grid(B / ROWS);                         // 32 workgroups
    dim3 block(NTHREADS);                        // 512 threads = 16 waves
    lstm_scan_wmma<<<grid, block, 0, stream>>>(x, W_ih, W_hh, b_ih, b_hh,
                                               W1, b1, W2, b2, out);
}
